// JointActor_90701119357586
// MI455X (gfx1250) — matrix operand order repaired and checked by reading, not compiled, not run on hardware
//
#include <hip/hip_runtime.h>
#include <stdint.h>

typedef __attribute__((ext_vector_type(16))) __bf16    v16bf;
typedef __attribute__((ext_vector_type(8)))  float     v8f;
typedef __attribute__((ext_vector_type(4)))  uint32_t  v4u;
typedef __attribute__((ext_vector_type(2)))  float     v2f;

#define DD        512            // feature depth
#define FTILES    32             // 512 / 16 output-feature tiles
#define KCHUNKS   16             // 512 / 32 reduction chunks
#define LDS_ROW   260            // 256 K-pair dwords + 4 pad (bank-conflict-free)

struct Frag { union { v4u q[2]; v16bf v; }; };

__device__ __forceinline__ uint32_t bf16_rne(float x) {
  uint32_t u = __float_as_uint(x);
  return (u + 0x7FFFu + ((u >> 16) & 1u)) >> 16;
}
__device__ __forceinline__ float bf16_f32(uint32_t h) { return __uint_as_float(h << 16); }

__device__ __forceinline__ void split_pack(float x0, float x1, uint32_t& hi, uint32_t& lo) {
  uint32_t h0 = bf16_rne(x0), h1 = bf16_rne(x1);
  uint32_t l0 = bf16_rne(x0 - bf16_f32(h0));
  uint32_t l1 = bf16_rne(x1 - bf16_f32(h1));
  hi = h0 | (h1 << 16);
  lo = l0 | (l1 << 16);
}

// Stage 16 rows x 512 f32 (contiguous) into LDS as bf16 hi/lo, one K-pair per dword.
__device__ __forceinline__ void stage16(const float* __restrict__ src,
                                        uint32_t* __restrict__ lHi,
                                        uint32_t* __restrict__ lLo, int lane) {
  const v2f* s2 = reinterpret_cast<const v2f*>(src);
#pragma unroll 8
  for (int i = 0; i < 128; ++i) {
    int idx = lane + 32 * i;               // 0..4095; row = idx>>8, pair = idx&255
    v2f x = s2[idx];                       // coalesced 256B per wave
    uint32_t hi, lo;
    split_pack(x.x, x.y, hi, lo);
    int o = (idx >> 8) * LDS_ROW + (idx & 255);
    lHi[o] = hi;
    lLo[o] = lo;
  }
}

// A-fragment (16x32 bf16) per ISA layout:
//  lanes 0-15 (row = lane): VGPR0-3 = K 0..7 (pairs p0..p0+3), VGPR4-7 = K16..23 (pairs p0+8..)
//  lanes 16-31: +4 pair offset (K8..15 / K24..31)
__device__ __forceinline__ void load_a(const uint32_t* __restrict__ lds, int lane,
                                       int kc, Frag& fr) {
  int base = (lane & 15) * LDS_ROW + kc * 16 + ((lane >> 4) << 2);   // 16B aligned
  const v4u* p = reinterpret_cast<const v4u*>(lds + base);
  fr.q[0] = p[0];        // ds_load_b128
  fr.q[1] = p[2];        // +8 dwords
}

// B-fragment: prepacked so each lane reads 8 consecutive dwords (2x b128, coalesced).
__device__ __forceinline__ void load_b(const uint32_t* __restrict__ hi,
                                       const uint32_t* __restrict__ lo,
                                       int ft, int kc, int lane, Frag& fh, Frag& fl) {
  int idx = (((ft * 16 + kc) * 32 + lane) << 1);   // v4u index
  const v4u* ph = reinterpret_cast<const v4u*>(hi);
  const v4u* pl = reinterpret_cast<const v4u*>(lo);
  fh.q[0] = ph[idx]; fh.q[1] = ph[idx + 1];
  fl.q[0] = pl[idx]; fl.q[1] = pl[idx + 1];
}

__device__ __forceinline__ v8f wmma_acc(const Frag& a, const Frag& b, v8f c) {
  return __builtin_amdgcn_wmma_f32_16x16x32_bf16(false, a.v, false, b.v,
                                                 (short)0, c, false, false);
}

// ---------------------------------------------------------------------------
// Prep 1: repack a 512x512 fp32 weight (row-major [d][f]) into B-fragment
// layout, bf16 hi/lo.  B layout (32x16, bf16): lanes 0-15 hold col n = lane,
// K=0..7 in VGPR0-3, K=16..23 in VGPR4-7 (2/VGPR); lanes 16-31 hold K=8..15 / K=24..31.
// Packed dword position = ((ft*16+kc)*32 + lane)*8 + j, pair p = kc*16 + (j>>2)*8 + (lane>>4)*4 + (j&3).
__global__ __launch_bounds__(256) void pack_w_kernel(const float* __restrict__ W,
                                                     uint32_t* __restrict__ hi,
                                                     uint32_t* __restrict__ lo) {
  int t = blockIdx.x * 256 + threadIdx.x;  // 0..131071
  int f = t & 511;
  int p = t >> 9;                          // global K-pair 0..255
  float x0 = W[(2 * p) * DD + f];          // coalesced over f
  float x1 = W[(2 * p + 1) * DD + f];
  uint32_t h, l;
  split_pack(x0, x1, h, l);
  int kc = p >> 4, rem = p & 15, hf = (rem >> 2) & 1, j = (rem & 3) | ((rem >> 3) << 2);
  int lane = hf * 16 + (f & 15);
  int ft = f >> 4;
  int pos = ((ft * 16 + kc) * 32 + lane) * 8 + j;
  hi[pos] = h;
  lo[pos] = l;
}

// Prep 2: biasB[b][f] = graph_emb[b]@Wg + loc_global_emb[b]@Wl + b1[f]   (8x512)
__global__ __launch_bounds__(64) void bias_kernel(const float* __restrict__ g,
                                                  const float* __restrict__ lg,
                                                  const float* __restrict__ W1,
                                                  const float* __restrict__ b1,
                                                  float* __restrict__ biasB) {
  int t = blockIdx.x * 64 + threadIdx.x;   // 0..4095
  int b = t >> 9, f = t & 511;
  const float* Wg = W1 + (size_t)1 * DD * DD;   // rows 512..1023
  const float* Wl = W1 + (size_t)2 * DD * DD;   // rows 1024..1535
  float acc = b1[f];
  for (int d = 0; d < DD; ++d) {
    acc = fmaf(g[b * DD + d], Wg[d * DD + f], acc);    // coalesced over f
    acc = fmaf(lg[b * DD + d], Wl[d * DD + f], acc);
  }
  biasB[t] = acc;
}

// ---------------------------------------------------------------------------
// Phase 1: nodePre[r][f] = node_embs[r]@Wn + biasB[b][f]   (r = b*N+n, 4096 rows)
__global__ __launch_bounds__(32) void node_gemm_kernel(const float* __restrict__ node,
                                                       const uint32_t* __restrict__ bHi,
                                                       const uint32_t* __restrict__ bLo,
                                                       const float* __restrict__ biasB,
                                                       float* __restrict__ nodePre) {
  __shared__ uint32_t lHi[16 * LDS_ROW];
  __shared__ uint32_t lLo[16 * LDS_ROW];
  int lane = threadIdx.x & 31;
  int tile = blockIdx.x;                   // 0..255, rows tile*16..tile*16+15
  stage16(node + (size_t)tile * 16 * DD, lHi, lLo, lane);
  __syncthreads();
  int bIdx = tile >> 5;                    // 32 tiles per batch (512 rows / 16)

  for (int ft = 0; ft < FTILES; ++ft) {
    v8f c = {};
#pragma unroll 2
    for (int kc = 0; kc < KCHUNKS; ++kc) {
      Frag aHi, aLo, wHi, wLo;
      load_a(lHi, lane, kc, aHi);
      load_a(lLo, lane, kc, aLo);
      load_b(bHi, bLo, ft, kc, lane, wHi, wLo);
      c = wmma_acc(aHi, wHi, c);
      c = wmma_acc(aHi, wLo, c);
      c = wmma_acc(aLo, wHi, c);
    }
    int f = ft * 16 + (lane & 15);
    float bias = biasB[bIdx * DD + f];
    int rbase = tile * 16 + ((lane >> 4) << 3);   // C layout: vgpr j -> row j (+8 hi half)
#pragma unroll
    for (int j = 0; j < 8; ++j)
      nodePre[(size_t)(rbase + j) * DD + f] = c[j] + bias;
  }
}

// ---------------------------------------------------------------------------
// Phase 2: per (b,n): 16x512 loc tile @ Wloc, + nodePre, ReLU, dot W2 -> 16 scores.
__global__ __launch_bounds__(32) void loc_score_kernel(const float* __restrict__ loc,
                                                       const uint32_t* __restrict__ bHi,
                                                       const uint32_t* __restrict__ bLo,
                                                       const float* __restrict__ nodePre,
                                                       const float* __restrict__ W2,
                                                       const float* __restrict__ b2,
                                                       float* __restrict__ out) {
  __shared__ uint32_t lHi[16 * LDS_ROW];
  __shared__ uint32_t lLo[16 * LDS_ROW];
  int lane = threadIdx.x & 31;
  int tile = blockIdx.x;                   // 0..4095 == b*N + n; M-rows are k=0..15
  stage16(loc + (size_t)tile * 16 * DD, lHi, lLo, lane);
  __syncthreads();

  float sacc[8] = {0.f, 0.f, 0.f, 0.f, 0.f, 0.f, 0.f, 0.f};

  for (int ft = 0; ft < FTILES; ++ft) {
    v8f c = {};
#pragma unroll 2
    for (int kc = 0; kc < KCHUNKS; ++kc) {
      Frag aHi, aLo, wHi, wLo;
      load_a(lHi, lane, kc, aHi);
      load_a(lLo, lane, kc, aLo);
      load_b(bHi, bLo, ft, kc, lane, wHi, wLo);
      c = wmma_acc(aHi, wHi, c);
      c = wmma_acc(aHi, wLo, c);
      c = wmma_acc(aLo, wHi, c);
    }
    int f = ft * 16 + (lane & 15);
    float np  = nodePre[(size_t)tile * DD + f];   // same (b,n) for all 16 rows
    float w2v = W2[f];
#pragma unroll
    for (int j = 0; j < 8; ++j) {
      float h = c[j] + np;                 // pre-activation for row (j + 8*(lane>>4)), col f
      h = h > 0.f ? h : 0.f;               // ReLU
      sacc[j] = fmaf(h, w2v, sacc[j]);     // partial dot with W2
    }
  }

  // Horizontal reduce across the 16 lanes of each half (feature dimension).
#pragma unroll
  for (int j = 0; j < 8; ++j) {
    float v = sacc[j];
    v += __shfl_xor(v, 8, 16);
    v += __shfl_xor(v, 4, 16);
    v += __shfl_xor(v, 2, 16);
    v += __shfl_xor(v, 1, 16);
    sacc[j] = v;
  }
  float bb = b2[0];
  if ((lane & 15) == 0) {
    int rowoff = (lane >> 4) << 3;         // lane 0 -> rows 0..7, lane 16 -> rows 8..15
#pragma unroll
    for (int j = 0; j < 8; ++j)
      out[tile * 16 + rowoff + j] = sacc[j] + bb;
  }
}

// ---------------------------------------------------------------------------
extern "C" void kernel_launch(void* const* d_in, const int* in_sizes, int n_in,
                              void* d_out, int out_size, void* d_ws, size_t ws_size,
                              hipStream_t stream) {
  const float* node = (const float*)d_in[0];   // (8,512,512)
  const float* loc  = (const float*)d_in[1];   // (8,512,16,512)
  const float* g    = (const float*)d_in[2];   // (8,512)
  const float* lg   = (const float*)d_in[3];   // (8,512)
  const float* W1   = (const float*)d_in[4];   // (2048,512)
  const float* b1   = (const float*)d_in[5];   // (512,)
  const float* W2   = (const float*)d_in[6];   // (512,1)
  const float* b2   = (const float*)d_in[7];   // (1,)
  float* out = (float*)d_out;                  // (8, 8192)

  // Workspace layout (~10.5 MB):
  char* ws = (char*)d_ws;
  float*    nodePre = (float*)ws;                               // 4096*512 f32 = 8 MB
  uint32_t* wnHi = (uint32_t*)(ws + (size_t)8 * 1024 * 1024);   // 512 KB each
  uint32_t* wnLo = wnHi + 131072;
  uint32_t* wlHi = wnLo + 131072;
  uint32_t* wlLo = wlHi + 131072;
  float*    biasB = (float*)(wlLo + 131072);                    // 8*512 f32

  // Prep: pack Wn (W1 rows 0..511) and Wloc (rows 1536..2047) into B-fragment bf16 hi/lo.
  pack_w_kernel<<<512, 256, 0, stream>>>(W1, wnHi, wnLo);
  pack_w_kernel<<<512, 256, 0, stream>>>(W1 + (size_t)3 * DD * DD, wlHi, wlLo);
  bias_kernel<<<64, 64, 0, stream>>>(g, lg, W1, b1, biasB);

  // Phase 1: node GEMM (+bias) -> nodePre.
  node_gemm_kernel<<<256, 32, 0, stream>>>(node, wnHi, wnLo, biasB, nodePre);

  // Phase 2: loc GEMM + nodePre + ReLU + dot(W2) + b2 -> scores.
  loc_score_kernel<<<4096, 32, 0, stream>>>(loc, wlHi, wlLo, nodePre, W2, b2, out);
}
